// DHCNN_83322365543015
// MI455X (gfx1250) — matrix operand order repaired and checked by reading, not compiled, not run on hardware
//
#include <hip/hip_runtime.h>

// ---------------------------------------------------------------------------
// DHCNN recurrence for MI455X (gfx1250, wave32, WMMA 16x16x32 f16)
//
// D=3 layers, T = 256 past + 64 forecast = 320 steps, NS=512, NY=32, B=256.
// Batch rows are independent -> 16 workgroups (16 rows each), no grid sync.
// Layers advance in lockstep per timestep -> no lower_states global buffer.
// ---------------------------------------------------------------------------

typedef _Float16 half8  __attribute__((ext_vector_type(8)));
typedef _Float16 half16 __attribute__((ext_vector_type(16)));
typedef float    f32x8  __attribute__((ext_vector_type(8)));

#define D_LAYERS 3
#define T_PAST   256
#define T_FC     64
#define T_TOT    (T_PAST + T_FC)
#define NS       512
#define NY       32
#define BATCH    256
#define MT       16                 // batch rows per workgroup == WMMA M
#define NBLK     (BATCH / MT)       // 16 workgroups
#define NT_PER_WAVE 4               // 8 waves * 4 tiles * 16 cols = 512 = NS

// LDS layout (dynamic shared memory)
#define SM_H_BYTES    (D_LAYERS * MT * NS * 4)   // f32 hidden state, 3 layers
#define SM_SLOW_BYTES (MT * NS * 4)              // f32 cascade (s_in of layer i)
#define SM_TS_BYTES   (MT * NS * 2)              // f16 tanh(s_corr)
#define SM_GATE_BYTES (NS * 4)                   // f32 clipped lstm gate
#define SM_SLOW_OFF   (SM_H_BYTES)
#define SM_TS_OFF     (SM_SLOW_OFF + SM_SLOW_BYTES)
#define SM_GATE_OFF   (SM_TS_OFF + SM_TS_BYTES)
#define SM_TOTAL      (SM_GATE_OFF + SM_GATE_BYTES)   // 149504 bytes

// --------------------------- weight conversion -----------------------------
__global__ void dhcnn_cvt_f16(const float* __restrict__ src,
                              _Float16* __restrict__ dst, int n) {
    int i = blockIdx.x * blockDim.x + threadIdx.x;
    if (i < n) dst[i] = (_Float16)src[i];
}

// ------------------------------ main kernel --------------------------------
__global__ __launch_bounds__(256) void dhcnn_main(
    const float* __restrict__ Y,           // (T_PAST, B, NY)
    const float* __restrict__ init_state,  // (D, 1, NS)
    const float* __restrict__ lstm_gate,   // (NS)
    const _Float16* __restrict__ wA,       // (D, NS, NS) f16, row-major
    float* __restrict__ out)               // (D, T_TOT, B, NY)
{
    extern __shared__ char raw[];
    float*    hL    = (float*)raw;                     // [D][MT][NS]
    float*    sLow  = (float*)(raw + SM_SLOW_OFF);     // [MT][NS]
    _Float16* tS    = (_Float16*)(raw + SM_TS_OFF);    // [MT][NS]
    float*    gateL = (float*)(raw + SM_GATE_OFF);     // [NS]

    const int tid     = threadIdx.x;
    const int rowBase = blockIdx.x * MT;
    const int lane    = tid & 31;
    const int wave    = tid >> 5;
    const int m       = lane & 15;
    const bool lo16   = lane < 16;

    // one-time init: clipped gate + broadcast initial hidden state
    for (int c = tid; c < NS; c += 256) {
        float g = lstm_gate[c];
        gateL[c] = fminf(fmaxf(g, 0.0f), 1.0f);
    }
    for (int e = tid; e < D_LAYERS * MT * NS; e += 256) {
        int i   = e / (MT * NS);
        int col = e & (NS - 1);
        hL[e] = init_state[i * NS + col];
    }
    __syncthreads();

    for (int t = 0; t < T_TOT; ++t) {
        const bool past = t < T_PAST;

        for (int i = 0; i < D_LAYERS; ++i) {
            // ---- elementwise: cascade s_in, emit err/fc, stage tanh(s_corr)
            for (int e = tid; e < MT * NS; e += 256) {
                const int row = e >> 9;          // NS == 512
                const int col = e & (NS - 1);
                float hv  = hL[i * MT * NS + e];
                float sIn = hv + (i == 0 ? 0.0f : sLow[e]);
                sLow[e] = sIn;                   // becomes lower for layer i+1
                float sc = sIn;
                if (col < NY) {
                    const int gr = rowBase + row;
                    float* op = out + (((size_t)i * T_TOT + t) * BATCH + gr) * NY + col;
                    if (past) {
                        float yv = Y[((size_t)t * BATCH + gr) * NY + col];
                        *op = sIn - yv;          // err
                        sc  = yv;                // s_corr[:, :NY] = Y[t]
                    } else {
                        *op = sIn;               // forecast
                    }
                }
                tS[e] = (_Float16)tanhf(sc);
            }
            __syncthreads();

            // ---- matmul: upd = tanh(s_corr) @ A_i^T  (16 x 512 x 512)
            const _Float16* wL = wA + (size_t)i * NS * NS;

            f32x8 acc[NT_PER_WAVE];
#pragma unroll
            for (int j = 0; j < NT_PER_WAVE; ++j) acc[j] = (f32x8)0.0f;

            for (int c = 0; c < NS / 32; ++c) {
                // A fragment (16x32 f16): lanes 0-15 -> K pairs {c*32+0..7, +16..23}
                //                         lanes 16-31 -> {c*32+8..15, +24..31}
                const int kA = c * 32 + (lo16 ? 0 : 8);
                half8 a0 = *(const half8*)&tS[m * NS + kA];
                half8 a1 = *(const half8*)&tS[m * NS + kA + 16];
                half16 afrag = __builtin_shufflevector(
                    a0, a1, 0, 1, 2, 3, 4, 5, 6, 7, 8, 9, 10, 11, 12, 13, 14, 15);

                // B fragment (32x16 f16): B[k,n] = A_w[n,k]; lane n = m,
                // lanes 0-15 hold K={c*32+0..7, +16..23}, lanes 16-31 hold K={+8..15, +24..31}.
                const int kB = c * 32 + (lo16 ? 0 : 8);
#pragma unroll
                for (int j = 0; j < NT_PER_WAVE; ++j) {
                    const int n = (wave * NT_PER_WAVE + j) * 16 + m;
                    const _Float16* pb = &wL[(size_t)n * NS + kB];
                    if (c + 1 < NS / 32) __builtin_prefetch(pb + 32, 0, 3);
                    half8 b0 = *(const half8*)pb;
                    half8 b1 = *(const half8*)(pb + 16);
                    half16 bfrag = __builtin_shufflevector(
                        b0, b1, 0, 1, 2, 3, 4, 5, 6, 7, 8, 9, 10, 11, 12, 13, 14, 15);

                    acc[j] = __builtin_amdgcn_wmma_f32_16x16x32_f16(
                        /*neg_a=*/false, afrag, /*neg_b=*/false, bfrag,
                        /*c_mod=*/(short)0, acc[j],
                        /*reuse_a=*/false, /*reuse_b=*/false);
                }
            }

            // ---- hidden-state update from C-fragment layout
            // C layout: lane<16 -> M=v, N=lane ; lane>=16 -> M=v+8, N=lane-16
#pragma unroll
            for (int j = 0; j < NT_PER_WAVE; ++j) {
                const int colB = (wave * NT_PER_WAVE + j) * 16 + m;
#pragma unroll
                for (int v = 0; v < 8; ++v) {
                    const int row = v + (lo16 ? 0 : 8);
                    float upd = acc[j][v];
                    float hnew;
                    if (i == 0) {
                        // LSTM blend: h = d*upd + (1-d)*s_corr
                        float sc;
                        if (past && colB < NY)
                            sc = Y[((size_t)t * BATCH + rowBase + row) * NY + colB];
                        else
                            sc = sLow[row * NS + colB];
                        float dg = gateL[colB];
                        hnew = dg * upd + (1.0f - dg) * sc;
                    } else {
                        hnew = upd;
                    }
                    hL[(i * MT + row) * NS + colB] = hnew;
                }
            }
            __syncthreads();
        }
    }
}

// ------------------------------- launcher ----------------------------------
extern "C" void kernel_launch(void* const* d_in, const int* in_sizes, int n_in,
                              void* d_out, int out_size, void* d_ws, size_t ws_size,
                              hipStream_t stream) {
    (void)in_sizes; (void)n_in; (void)out_size; (void)ws_size;

    const float* Y          = (const float*)d_in[0];   // (256,256,32)
    const float* init_state = (const float*)d_in[1];   // (3,1,512)
    const float* A          = (const float*)d_in[2];   // (3,512,512)
    const float* lstm_gate  = (const float*)d_in[3];   // (512)
    // d_in[4] = forecast_horizon (64), compile-time constant here

    _Float16* wA  = (_Float16*)d_ws;                   // 1.5 MB f16 weights
    float*    out = (float*)d_out;

    const int nW = D_LAYERS * NS * NS;                 // 786432
    dhcnn_cvt_f16<<<dim3((nW + 255) / 256), dim3(256), 0, stream>>>(A, wA, nW);

    hipFuncSetAttribute((const void*)dhcnn_main,
                        hipFuncAttributeMaxDynamicSharedMemorySize, SM_TOTAL);

    dhcnn_main<<<dim3(NBLK), dim3(256), SM_TOTAL, stream>>>(
        Y, init_state, lstm_gate, wA, out);
}
